// SNNPolicy_48284022342088
// MI455X (gfx1250) — matrix operand order repaired and checked by reading, not compiled, not run on hardware
//
#include <hip/hip_runtime.h>
#include <cstdint>
#include <cstddef>

// ---------------------------------------------------------------------------
// SNN (2x recurrent LIF layers) on gfx1250 via bf16 WMMA (f32 accumulate).
// B=4096, D=512, H=1024, T=16.  D = A(bf16) x B(bf16) + C(f32) with
// v_wmma_f32_16x16x32_bf16.  Weights pre-transposed to K-major bf16 so the
// W tile stages as contiguous 16B rows; WMMA B-operand (lane = N, elems = K) is gathered per element.
// Double-buffered LDS with GLOBAL_LOAD_ASYNC_TO_LDS_B128 (ASYNCcnt) so the
// next tile's global->LDS DMA overlaps the current tile's WMMAs.
// ---------------------------------------------------------------------------

typedef __attribute__((ext_vector_type(16))) __bf16 v16bf;
typedef __attribute__((ext_vector_type(8)))  float  v8f;

// Types matching __builtin_amdgcn_global_load_async_to_lds_b128's params:
//   (v4i addrspace(1)*, v4i addrspace(3)*, imm int offset, imm int cpol)
typedef int v4i __attribute__((vector_size(16)));
typedef __attribute__((address_space(1))) v4i gv4i;
typedef __attribute__((address_space(3))) v4i sv4i;

#define B_DIM   4096
#define D_DIM   512
#define H_DIM   1024
#define T_STEPS 16

#define BM 128
#define BN 128
#define BK 32
#define ASTR 40    // padded LDS row stride (bf16 elems) for A tile 128x32
#define BSTR 136   // padded LDS row stride (bf16 elems) for W tile  32x128

#if defined(__has_builtin)
#  if __has_builtin(__builtin_amdgcn_global_load_async_to_lds_b128)
#    define HAVE_ASYNC_LDS 1
#  else
#    define HAVE_ASYNC_LDS 0
#  endif
#else
#  define HAVE_ASYNC_LDS 0
#endif

__device__ __forceinline__ void async_wait0() {
#if HAVE_ASYNC_LDS
#  if __has_builtin(__builtin_amdgcn_s_wait_asynccnt)
  __builtin_amdgcn_s_wait_asynccnt(0);
#  else
  asm volatile("s_wait_asynccnt 0x0" ::: "memory");
#  endif
#endif
}

// ---- bf16 helpers (bit-level, RNE) ----------------------------------------
__device__ __forceinline__ unsigned short f2bf(float f) {
  unsigned int u = __float_as_uint(f);
  unsigned int r = u + 0x7FFFu + ((u >> 16) & 1u);
  return (unsigned short)(r >> 16);
}

// ---- utility kernels -------------------------------------------------------
__global__ void zero_u32_kernel(unsigned int* __restrict__ p, size_t n) {
  size_t i = (size_t)blockIdx.x * blockDim.x + threadIdx.x;
  size_t stride = (size_t)gridDim.x * blockDim.x;
  for (; i < n; i += stride) p[i] = 0u;
}

__global__ void conv_f2bf_kernel(const float* __restrict__ src,
                                 unsigned short* __restrict__ dst, size_t n) {
  size_t i = (size_t)blockIdx.x * blockDim.x + threadIdx.x;
  if (i < n) dst[i] = f2bf(src[i]);
}

// Wt[c*R + r] = bf16(W[r*C + c])  (row-major (R,C) -> K-major (C,R))
__global__ void transpose_f2bf_kernel(const float* __restrict__ W,
                                      unsigned short* __restrict__ Wt,
                                      int R, int C) {
  size_t i = (size_t)blockIdx.x * blockDim.x + threadIdx.x;
  if (i < (size_t)R * C) {
    int r = (int)(i / C);
    int c = (int)(i % C);
    Wt[(size_t)c * R + r] = f2bf(W[i]);
  }
}

__global__ void finalize_kernel(const unsigned int* __restrict__ cnt,
                                float* __restrict__ out) {
  out[(size_t)B_DIM * H_DIM] =
      (float)(*cnt) * (1.0f / ((float)B_DIM * (float)H_DIM * (float)T_STEPS));
}

// ---------------------------------------------------------------------------
// Fused GEMM (+ optional second GEMM summed) + epilogue.
//   acc = A1 @ W1t  (+ A2 @ W2t)      A: (M x K) bf16 row-major, ld = K
//                                     Wt: (K x N) bf16 row-major, ld = N
//   mode 0 (FC):  out_bf = bf16(acc + bias[col])
//   mode 1 (LIF): v_dec = v + 0.1*(i - v); z = (v_dec > 1)
//                 v' = (1-z)*v_dec;  i' = 0.8*i + acc
//                 out_bf = bf16(z); optional f32 z to zf32; spike count to cnt
// 256 threads = 8 waves; 128x128 block tile; each wave: 4x2 subtiles of 16x16.
// ---------------------------------------------------------------------------
__launch_bounds__(256)
__global__ void gemm_lif_kernel(
    const unsigned short* __restrict__ A1, const unsigned short* __restrict__ W1, int K1,
    const unsigned short* __restrict__ A2, const unsigned short* __restrict__ W2, int K2,
    const float* __restrict__ bias, int N, int mode,
    unsigned short* __restrict__ out_bf,
    float* __restrict__ vbuf, float* __restrict__ ibuf,
    float* __restrict__ zf32, unsigned int* __restrict__ cnt) {

  __shared__ __align__(16) unsigned short Alds[2][BM * ASTR];
  __shared__ __align__(16) unsigned short Blds[2][BK * BSTR];
  __shared__ unsigned int scnt;

  const int tid    = threadIdx.x;
  const int lane   = tid & 31;
  const int wave   = tid >> 5;
  const int wave_m = wave >> 2;   // 0..1 -> 64-row half
  const int wave_n = wave & 3;    // 0..3 -> 32-col quarter
  const int bm0    = blockIdx.y * BM;
  const int bn0    = blockIdx.x * BN;

  union FragU { uint4 u[2]; v16bf v; };

  const v8f z8 = {0.f, 0.f, 0.f, 0.f, 0.f, 0.f, 0.f, 0.f};
  v8f acc[8];
#pragma unroll
  for (int s = 0; s < 8; ++s) acc[s] = z8;

  // K split across lane halves per ISA A-matrix layout (16-bit, 16x32)
  const int kg = (lane < 16) ? 0 : 8;
  const int lq = lane & 15;

  const int nt1 = K1 / BK;
  const int nt2 = (A2 != nullptr) ? (K2 / BK) : 0;
  const int nt  = nt1 + nt2;

  // Stage virtual K-tile t into LDS buffer `buf` (4 x 16B per thread).
  auto stage = [&](int t, int buf) {
    const unsigned short* Ag;
    const unsigned short* Wg;
    int ldA, k0;
    if (t < nt1) { Ag = A1; Wg = W1; ldA = K1; k0 = t * BK; }
    else         { Ag = A2; Wg = W2; ldA = K2; k0 = (t - nt1) * BK; }
    unsigned short* Al = &Alds[buf][0];
    unsigned short* Bl = &Blds[buf][0];
#pragma unroll
    for (int it = 0; it < 2; ++it) {
      int c  = tid + it * 256;           // 0..511
      int ar = c >> 2;                   // A: row 0..127
      int ac = (c & 3) << 3;             // A: col 0,8,16,24
      int wr = c >> 4;                   // W: row 0..31
      int wc = (c & 15) << 3;            // W: col 0..120
      const unsigned short* gA = &Ag[(size_t)(bm0 + ar) * ldA + (k0 + ac)];
      const unsigned short* gB = &Wg[(size_t)(k0 + wr) * N + (bn0 + wc)];
      unsigned short* lA = &Al[ar * ASTR + ac];
      unsigned short* lB = &Bl[wr * BSTR + wc];
#if HAVE_ASYNC_LDS
      __builtin_amdgcn_global_load_async_to_lds_b128((gv4i*)gA, (sv4i*)lA, 0, 0);
      __builtin_amdgcn_global_load_async_to_lds_b128((gv4i*)gB, (sv4i*)lB, 0, 0);
#else
      *(uint4*)lA = *(const uint4*)gA;
      *(uint4*)lB = *(const uint4*)gB;
#endif
    }
  };

  if (nt > 0) stage(0, 0);

#pragma unroll 1
  for (int t = 0; t < nt; ++t) {
    const int cur = t & 1;
    async_wait0();        // current buffer's DMA complete (no-op w/o async)
    __syncthreads();      // visible to all waves; prev compute done
    if (t + 1 < nt) stage(t + 1, cur ^ 1);   // overlap next DMA with WMMAs

    // ---- build fragments from LDS ----
    FragU fu;
    v16bf afr[4], bfr[2];
    const unsigned short* Al = &Alds[cur][0];
    const unsigned short* Bl = &Blds[cur][0];
#pragma unroll
    for (int sm = 0; sm < 4; ++sm) {
      int rr = (wave_m * 64 + sm * 16 + lq) * ASTR;
      fu.u[0] = *(const uint4*)&Al[rr + kg];        // K = kg .. kg+7
      fu.u[1] = *(const uint4*)&Al[rr + kg + 16];   // K = kg+16 .. kg+23
      afr[sm] = fu.v;
    }
#pragma unroll
    for (int sn = 0; sn < 2; ++sn) {
      int cc = wave_n * 32 + sn * 16 + lq; // lane = N, elems = K (table order)
#pragma unroll
      for (int s = 0; s < 16; ++s)
        ((unsigned short*)fu.u)[s] = Bl[((s < 8) ? (kg + s) : (kg + 8 + s)) * BSTR + cc];
      bfr[sn] = fu.v;
    }

    // ---- 8 WMMAs ----
#pragma unroll
    for (int sm = 0; sm < 4; ++sm)
#pragma unroll
      for (int sn = 0; sn < 2; ++sn)
        acc[sm * 2 + sn] = __builtin_amdgcn_wmma_f32_16x16x32_bf16(
            false, afr[sm], false, bfr[sn], (short)0, acc[sm * 2 + sn],
            false, false);
  }

  // ---- epilogue: C/D layout -> lane lq = N, reg r = M (lanes>=16: M=r+8) ---
  const int mhalf = (lane >> 4) << 3;
  unsigned int lc = 0;
#pragma unroll
  for (int st = 0; st < 8; ++st) {
    int sm = st >> 1, sn = st & 1;
    int col = bn0 + wave_n * 32 + sn * 16 + lq;
    int rowb = bm0 + wave_m * 64 + sm * 16 + mhalf;
    float bc = (bias != nullptr) ? bias[col] : 0.0f;
#pragma unroll
    for (int r = 0; r < 8; ++r) {
      size_t idx = (size_t)(rowb + r) * N + col;
      float a = acc[st][r];
      if (mode == 0) {
        out_bf[idx] = f2bf(a + bc);
      } else {
        float vv = vbuf[idx];
        float ii = ibuf[idx];
        float vdec = vv + 0.1f * (ii - vv);       // dt*tau_mem_inv = 0.1
        float zn = (vdec > 1.0f) ? 1.0f : 0.0f;   // v_th = 1
        vbuf[idx] = (1.0f - zn) * vdec;           // v_reset = 0
        ibuf[idx] = 0.8f * ii + a;                // (1 - dt*tau_syn_inv)
        out_bf[idx] = (zn > 0.0f) ? (unsigned short)0x3F80u : (unsigned short)0u;
        if (zf32) zf32[idx] = zn;
        lc += (zn > 0.0f) ? 1u : 0u;
      }
    }
  }

  if (mode == 1 && cnt != nullptr) {
    if (tid == 0) scnt = 0u;
    __syncthreads();
    atomicAdd(&scnt, lc);
    __syncthreads();
    if (tid == 0) atomicAdd(cnt, scnt);
  }
}

// ---------------------------------------------------------------------------
extern "C" void kernel_launch(void* const* d_in, const int* in_sizes, int n_in,
                              void* d_out, int out_size, void* d_ws, size_t ws_size,
                              hipStream_t stream) {
  (void)in_sizes; (void)n_in; (void)out_size; (void)ws_size;

  const float* x     = (const float*)d_in[0];
  const float* fc1_w = (const float*)d_in[1];   // (H, D)
  const float* fc1_b = (const float*)d_in[2];
  const float* w1in  = (const float*)d_in[3];   // (H, H)
  const float* w1rec = (const float*)d_in[4];
  const float* fc2_w = (const float*)d_in[5];
  const float* fc2_b = (const float*)d_in[6];
  const float* w2in  = (const float*)d_in[7];
  const float* w2rec = (const float*)d_in[8];
  float* out = (float*)d_out;

  const size_t BH = (size_t)B_DIM * H_DIM;
  char* ws = (char*)d_ws;
  size_t off = 0;
  auto alloc = [&](size_t bytes) -> char* {
    char* p = ws + off;
    off += (bytes + 255) & ~(size_t)255;
    return p;
  };

  // Buffers that must start as zero come first (one zero kernel covers them).
  float* v1 = (float*)alloc(BH * 4);
  float* i1 = (float*)alloc(BH * 4);
  float* v2 = (float*)alloc(BH * 4);
  float* i2 = (float*)alloc(BH * 4);
  unsigned short* z1a = (unsigned short*)alloc(BH * 2);
  unsigned short* z2a = (unsigned short*)alloc(BH * 2);
  unsigned int* cnt = (unsigned int*)alloc(256);
  const size_t zero_words = off / 4;

  unsigned short* z1b   = (unsigned short*)alloc(BH * 2);
  unsigned short* z2b   = (unsigned short*)alloc(BH * 2);
  unsigned short* z0    = (unsigned short*)alloc(BH * 2);
  unsigned short* z2in  = (unsigned short*)alloc(BH * 2);
  unsigned short* xb    = (unsigned short*)alloc((size_t)B_DIM * D_DIM * 2);
  unsigned short* fc1wt = (unsigned short*)alloc((size_t)D_DIM * H_DIM * 2);
  unsigned short* fc2wt = (unsigned short*)alloc((size_t)H_DIM * H_DIM * 2);
  unsigned short* w1it  = (unsigned short*)alloc((size_t)H_DIM * H_DIM * 2);
  unsigned short* w1rt  = (unsigned short*)alloc((size_t)H_DIM * H_DIM * 2);
  unsigned short* w2it  = (unsigned short*)alloc((size_t)H_DIM * H_DIM * 2);
  unsigned short* w2rt  = (unsigned short*)alloc((size_t)H_DIM * H_DIM * 2);

  // ---- prep ----
  zero_u32_kernel<<<8192, 256, 0, stream>>>((unsigned int*)ws, zero_words);

  {
    size_t n = (size_t)B_DIM * D_DIM;
    conv_f2bf_kernel<<<(unsigned)((n + 255) / 256), 256, 0, stream>>>(x, xb, n);
  }
  transpose_f2bf_kernel<<<(H_DIM * D_DIM + 255) / 256, 256, 0, stream>>>(fc1_w, fc1wt, H_DIM, D_DIM);
  transpose_f2bf_kernel<<<(H_DIM * H_DIM + 255) / 256, 256, 0, stream>>>(fc2_w, fc2wt, H_DIM, H_DIM);
  transpose_f2bf_kernel<<<(H_DIM * H_DIM + 255) / 256, 256, 0, stream>>>(w1in,  w1it,  H_DIM, H_DIM);
  transpose_f2bf_kernel<<<(H_DIM * H_DIM + 255) / 256, 256, 0, stream>>>(w1rec, w1rt,  H_DIM, H_DIM);
  transpose_f2bf_kernel<<<(H_DIM * H_DIM + 255) / 256, 256, 0, stream>>>(w2in,  w2it,  H_DIM, H_DIM);
  transpose_f2bf_kernel<<<(H_DIM * H_DIM + 255) / 256, 256, 0, stream>>>(w2rec, w2rt,  H_DIM, H_DIM);

  dim3 grid(H_DIM / BN, B_DIM / BM);   // (8, 32)
  dim3 blk(256);

  // z0 = x @ fc1_w.T + fc1_b   (time-invariant)
  gemm_lif_kernel<<<grid, blk, 0, stream>>>(
      xb, fc1wt, D_DIM, nullptr, nullptr, 0,
      fc1_b, H_DIM, /*mode=*/0, z0, nullptr, nullptr, nullptr, nullptr);

  unsigned short* z1buf[2] = {z1a, z1b};
  unsigned short* z2buf[2] = {z2a, z2b};

  for (int t = 0; t < T_STEPS; ++t) {
    unsigned short* z1r = z1buf[t & 1];
    unsigned short* z1w = z1buf[(t + 1) & 1];
    unsigned short* z2r = z2buf[t & 1];
    unsigned short* z2w = z2buf[(t + 1) & 1];

    // LIF1: i1' = 0.8*i1 + z0 @ w1in.T + z1_old @ w1rec.T  (+ spike/reset)
    gemm_lif_kernel<<<grid, blk, 0, stream>>>(
        z0, w1it, H_DIM, z1r, w1rt, H_DIM,
        nullptr, H_DIM, /*mode=*/1, z1w, v1, i1, nullptr, nullptr);

    // z2_in = z1_new @ fc2_w.T + fc2_b
    gemm_lif_kernel<<<grid, blk, 0, stream>>>(
        z1w, fc2wt, H_DIM, nullptr, nullptr, 0,
        fc2_b, H_DIM, /*mode=*/0, z2in, nullptr, nullptr, nullptr, nullptr);

    // LIF2: i2' = 0.8*i2 + z2_in @ w2in.T + z2_old @ w2rec.T  (+ count)
    gemm_lif_kernel<<<grid, blk, 0, stream>>>(
        z2in, w2it, H_DIM, z2r, w2rt, H_DIM,
        nullptr, H_DIM, /*mode=*/1, z2w, v2, i2,
        (t == T_STEPS - 1) ? out : nullptr, cnt);
  }

  finalize_kernel<<<1, 1, 0, stream>>>(cnt, out);
}
